// SimpleHGN_7868380086414
// MI455X (gfx1250) — compile-verified, matrix operand order checked by reading, not run on hardware
//
#include <hip/hip_runtime.h>
#include <hip/hip_bf16.h>
#include <math.h>

// ---- problem constants (match reference) ----
#define D_IN    768
#define HD      128
#define REL_DIM 32
#define N_REL   3
#define BETA    0.05f
#define NEG_SLOPE 0.2f
#define NT      (HD / 16)     // 8 column tiles (all GEMMs have Nc == 128)

typedef __attribute__((ext_vector_type(16))) __bf16 v16bf;
typedef __attribute__((ext_vector_type(8)))  float  v8f;

// -------------------------------------------------------------------------
// Pack B[K x 128] (f32) into per-lane WMMA bf16 fragments:
//   Bp[((s*NT + t)*32 + lane)] = 16 bf16 covering (k-step s, col tile t)
// Lane layout matches v_wmma_f32_16x16x32_bf16 B operand:
//   lane<16: col = t*16+lane, K = {k0..k0+7, k0+16..k0+23}
//   lane>=16: same col group, K = {k0+8..k0+15, k0+24..k0+31}
// -------------------------------------------------------------------------
__global__ void k_pack_b(const float* __restrict__ B, v16bf* __restrict__ Bp, int K)
{
    const int total = (K >> 5) * NT * 32;
    int idx = blockIdx.x * blockDim.x + threadIdx.x;
    if (idx >= total) return;
    const int lane = idx & 31;
    const int t = (idx >> 5) % NT;
    const int s = (idx >> 5) / NT;
    const int r  = lane & 15;
    const int kh = lane >> 4;
    const int col = t * 16 + r;
    const int k0 = s * 32 + kh * 8;
    v16bf b;
#pragma unroll
    for (int i = 0; i < 8; ++i) {
        b[i]     = (__bf16)B[(size_t)(k0 + i) * HD + col];
        b[8 + i] = (__bf16)B[(size_t)(k0 + 16 + i) * HD + col];
    }
    Bp[idx] = b;
}

// -------------------------------------------------------------------------
// Stripe GEMM: C[M x 128] = A[M x K] * B[K x 128], B pre-packed bf16 frags.
// One wave32 computes a full 16-row x 128-col stripe: A fragment loaded once
// per K-step (f32 -> bf16 in regs), reused by 8 WMMAs into 8 accumulators.
// A is read exactly once from HBM. Requires M%16==0, K%32==0.
// act: 0 = none, 1 = relu.  bias may be nullptr.
// -------------------------------------------------------------------------
__global__ void k_gemm_stripe_wmma(const float* __restrict__ A,
                                   const v16bf* __restrict__ Bp,
                                   const float* __restrict__ bias,
                                   float* __restrict__ C,
                                   int M, int K, int act)
{
    const int lane = threadIdx.x & 31;
    const int stripe = blockIdx.x * (blockDim.x >> 5) + (threadIdx.x >> 5);
    if (stripe >= (M >> 4)) return;
    const int r  = lane & 15;
    const int kh = lane >> 4;
    const int arow = stripe * 16 + r;

    v8f acc[NT] = {};
    const int nsteps = K >> 5;
    for (int s = 0; s < nsteps; ++s) {
        const int ka = (s << 5) + kh * 8;
        v16bf a;
        const float* ap = A + (size_t)arow * K + ka;
#pragma unroll
        for (int i = 0; i < 8; ++i) {
            a[i]     = (__bf16)ap[i];
            a[8 + i] = (__bf16)ap[16 + i];
        }
        const v16bf* bp = Bp + (size_t)s * (NT * 32) + lane;
#pragma unroll
        for (int t = 0; t < NT; ++t) {
            v16bf b = bp[t * 32];
            acc[t] = __builtin_amdgcn_wmma_f32_16x16x32_bf16(
                         false, a, false, b, (short)0, acc[t], false, false);
        }
    }

    // C/D layout: lanes 0-15 -> rows stripe*16 + v, lanes 16-31 -> + 8 + v
    const int crow0 = stripe * 16 + kh * 8;
#pragma unroll
    for (int t = 0; t < NT; ++t) {
        const int col = t * 16 + r;
        const float bv = bias ? bias[col] : 0.0f;
#pragma unroll
        for (int v = 0; v < 8; ++v) {
            float x = acc[t][v] + bv;
            if (act == 1) x = fmaxf(x, 0.0f);
            C[(size_t)(crow0 + v) * HD + col] = x;
        }
    }
}

// ---- relation precompute: rscal[k] = (rel[k] @ Wr) . a_r  (3 scalars) ----
__global__ void k_rel(const float* __restrict__ rel, const float* __restrict__ Wr,
                      const float* __restrict__ a, float* __restrict__ rscal)
{
    __shared__ float sh[N_REL][HD];
    const int hcol = threadIdx.x;           // blockDim.x == HD
    for (int k = 0; k < N_REL; ++k) {
        float s = 0.0f;
        for (int d = 0; d < REL_DIM; ++d)
            s += rel[k * REL_DIM + d] * Wr[d * HD + hcol];
        sh[k][hcol] = s;
    }
    __syncthreads();
    if (hcol < N_REL) {
        const float* ar = a + 2 * HD;
        float s = 0.0f;
        for (int j = 0; j < HD; ++j) s += sh[hcol][j] * ar[j];
        rscal[hcol] = s;
    }
}

// ---- per-node attention scores: ni = h.a_i, nj = h.a_j (wave per node) ----
__global__ void k_node_scores(const float* __restrict__ h, const float* __restrict__ a,
                              float* __restrict__ ni, float* __restrict__ nj, int n)
{
    const int lane = threadIdx.x & 31;
    const int node = (blockIdx.x * blockDim.x + threadIdx.x) >> 5;
    if (node >= n) return;
    const float* hp = h + (size_t)node * HD;
    float s0 = 0.0f, s1 = 0.0f;
#pragma unroll
    for (int j = lane; j < HD; j += 32) {
        float v = hp[j];
        s0 += v * a[j];
        s1 += v * a[HD + j];
    }
    for (int off = 16; off; off >>= 1) {
        s0 += __shfl_xor(s0, off, 32);
        s1 += __shfl_xor(s1, off, 32);
    }
    if (lane == 0) { ni[node] = s0; nj[node] = s1; }
}

__global__ void k_fill(float* __restrict__ p, float v, size_t n)
{
    size_t i = (size_t)blockIdx.x * blockDim.x + threadIdx.x;
    size_t stride = (size_t)gridDim.x * blockDim.x;
    for (; i < n; i += stride) p[i] = v;
}

// sign-split float atomic max (buffer initialized to -inf)
__device__ __forceinline__ void atomicMaxF(float* addr, float val)
{
    if (!(val == val)) return;
    if (val >= 0.0f) atomicMax((int*)addr, __float_as_int(val));
    else             atomicMin((unsigned int*)addr, __float_as_uint(val));
}

__global__ void k_edge_logit(const float* __restrict__ ni, const float* __restrict__ nj,
                             const float* __restrict__ rscal,
                             const int* __restrict__ src, const int* __restrict__ dst,
                             const int* __restrict__ et,
                             float* __restrict__ logit, float* __restrict__ maxb, int e_n)
{
    int e = blockIdx.x * blockDim.x + threadIdx.x;
    if (e >= e_n) return;
    int d = dst[e];
    float l = ni[d] + nj[src[e]] + rscal[et[e]];
    l = l > 0.0f ? l : NEG_SLOPE * l;
    logit[e] = l;
    atomicMaxF(maxb + d, l);
}

__global__ void k_edge_exp(const float* __restrict__ logit, const float* __restrict__ maxb,
                           const int* __restrict__ dst,
                           float* __restrict__ expv, float* __restrict__ sumb, int e_n)
{
    int e = blockIdx.x * blockDim.x + threadIdx.x;
    if (e >= e_n) return;
    int d = dst[e];
    float v = __expf(logit[e] - maxb[d]);
    expv[e] = v;
    atomicAdd(sumb + d, v);
}

__global__ void k_edge_alpha(const float* __restrict__ expv, const float* __restrict__ sumb,
                             const int* __restrict__ dst,
                             float* __restrict__ alpha, const float* __restrict__ pre, int e_n)
{
    int e = blockIdx.x * blockDim.x + threadIdx.x;
    if (e >= e_n) return;
    float al = expv[e] / (sumb[dst[e]] + 1e-16f);
    if (pre) al = al * (1.0f - BETA) + pre[e] * BETA;
    alpha[e] = al;
}

// ---- message aggregation: agg[dst] += h[src] * alpha  (wave per edge) ----
__global__ void k_aggregate(const float* __restrict__ h, const float* __restrict__ alpha,
                            const int* __restrict__ src, const int* __restrict__ dst,
                            float* __restrict__ agg, int e_n)
{
    const int lane = threadIdx.x & 31;
    const int e = (blockIdx.x * blockDim.x + threadIdx.x) >> 5;
    if (e >= e_n) return;
    const int s = src[e], d = dst[e];
    const float al = alpha[e];
    const float* hp = h + (size_t)s * HD;
    float* op = agg + (size_t)d * HD;
#pragma unroll
    for (int j = lane; j < HD; j += 32)
        atomicAdd(op + j, hp[j] * al);
}

__global__ void k_elu(const float* __restrict__ in, float* __restrict__ out, size_t n)
{
    size_t i = (size_t)blockIdx.x * blockDim.x + threadIdx.x;
    size_t stride = (size_t)gridDim.x * blockDim.x;
    for (; i < n; i += stride) {
        float x = in[i];
        out[i] = x > 0.0f ? x : expm1f(x);
    }
}

// ---- final head: out = pooled @ out_w + out_b  (wave per node, 2 outputs) ----
__global__ void k_head(const float* __restrict__ pooled, const float* __restrict__ ow,
                       const float* __restrict__ ob, float* __restrict__ out, int n)
{
    const int lane = threadIdx.x & 31;
    const int node = (blockIdx.x * blockDim.x + threadIdx.x) >> 5;
    if (node >= n) return;
    const float* pp = pooled + (size_t)node * HD;
    float s0 = 0.0f, s1 = 0.0f;
#pragma unroll
    for (int j = lane; j < HD; j += 32) {
        float v = pp[j];
        s0 += v * ow[j * 2 + 0];
        s1 += v * ow[j * 2 + 1];
    }
    for (int off = 16; off; off >>= 1) {
        s0 += __shfl_xor(s0, off, 32);
        s1 += __shfl_xor(s1, off, 32);
    }
    if (lane == 0) {
        out[node * 2 + 0] = s0 + ob[0];
        out[node * 2 + 1] = s1 + ob[1];
    }
}

// =========================================================================
extern "C" void kernel_launch(void* const* d_in, const int* in_sizes, int n_in,
                              void* d_out, int out_size, void* d_ws, size_t ws_size,
                              hipStream_t stream)
{
    const float* x      = (const float*)d_in[0];
    const int*   ei     = (const int*)  d_in[1];   // [2, E] flattened
    const int*   et     = (const int*)  d_in[2];
    const float* W0     = (const float*)d_in[3];
    const float* Wr0    = (const float*)d_in[4];
    const float* a0     = (const float*)d_in[5];
    const float* rel0   = (const float*)d_in[6];
    const float* W1     = (const float*)d_in[7];
    const float* Wr1    = (const float*)d_in[8];
    const float* a1     = (const float*)d_in[9];
    const float* rel1   = (const float*)d_in[10];
    const float* pool_w = (const float*)d_in[11];
    const float* pool_b = (const float*)d_in[12];
    const float* out_w  = (const float*)d_in[13];
    const float* out_b  = (const float*)d_in[14];
    float* out = (float*)d_out;

    const int n  = in_sizes[0] / D_IN;     // 50000
    const int en = in_sizes[2];            // 800000
    const int* src = ei;
    const int* dst = ei + en;

    // ---- workspace layout (f32 units; all offsets kept 32B aligned) ----
    float* w = (float*)d_ws;
    size_t off = 0;
    // packed-B fragment buffer (largest user: K=768 -> 24*8*32 frags * 8 f32)
    v16bf* Bp = (v16bf*)(w + off); off += (size_t)(D_IN / 32) * NT * 32 * 8;
    float* hbuf   = w + off; off += (size_t)n * HD;   // transformed features
    float* aggbuf = w + off; off += (size_t)n * HD;   // segment-sum accumulator
    float* xbuf   = w + off; off += (size_t)n * HD;   // elu output / next input
    float* pooled = w + off; off += (size_t)n * HD;
    float* ni     = w + off; off += (size_t)n;
    float* nj     = w + off; off += (size_t)n;
    float* maxb   = w + off; off += (size_t)n;
    float* sumb   = w + off; off += (size_t)n;
    float* rscal  = w + off; off += 8;                // padded for alignment
    float* logit  = w + off; off += (size_t)en;
    float* expv   = w + off; off += (size_t)en;
    float* alpha0 = w + off; off += (size_t)en;
    float* alpha1 = w + off; off += (size_t)en;
    (void)ws_size; (void)n_in; (void)out_size;

    const size_t nh = (size_t)n * HD;
    const int TPB = 256;
    const int wpb = TPB / 32;
    const int edge_blk  = (en + TPB - 1) / TPB;
    const int edgew_blk = (en + wpb - 1) / wpb;
    const int nodew_blk = (n  + wpb - 1) / wpb;
    const int fill_blk  = 2048;

    auto gemm = [&](const float* A, const float* B, const float* bias,
                    float* C, int M, int K, int act) {
        int pk_total = (K >> 5) * NT * 32;
        k_pack_b<<<(pk_total + TPB - 1) / TPB, TPB, 0, stream>>>(B, Bp, K);
        int stripes = M >> 4;
        k_gemm_stripe_wmma<<<(stripes + wpb - 1) / wpb, TPB, 0, stream>>>(
            A, Bp, bias, C, M, K, act);
    };

    // ================= Layer 0 =================
    k_rel<<<1, HD, 0, stream>>>(rel0, Wr0, a0, rscal);
    gemm(x, W0, nullptr, hbuf, n, D_IN, 0);
    k_node_scores<<<nodew_blk, TPB, 0, stream>>>(hbuf, a0, ni, nj, n);
    k_fill<<<fill_blk, TPB, 0, stream>>>(maxb, -INFINITY, n);
    k_fill<<<fill_blk, TPB, 0, stream>>>(sumb, 0.0f, n);
    k_fill<<<fill_blk, TPB, 0, stream>>>(aggbuf, 0.0f, nh);
    k_edge_logit<<<edge_blk, TPB, 0, stream>>>(ni, nj, rscal, src, dst, et, logit, maxb, en);
    k_edge_exp<<<edge_blk, TPB, 0, stream>>>(logit, maxb, dst, expv, sumb, en);
    k_edge_alpha<<<edge_blk, TPB, 0, stream>>>(expv, sumb, dst, alpha0, nullptr, en);
    k_aggregate<<<edgew_blk, TPB, 0, stream>>>(hbuf, alpha0, src, dst, aggbuf, en);
    k_elu<<<fill_blk, TPB, 0, stream>>>(aggbuf, xbuf, nh);

    // ================= Layer 1 =================
    k_rel<<<1, HD, 0, stream>>>(rel1, Wr1, a1, rscal);
    gemm(xbuf, W1, nullptr, hbuf, n, HD, 0);
    k_node_scores<<<nodew_blk, TPB, 0, stream>>>(hbuf, a1, ni, nj, n);
    k_fill<<<fill_blk, TPB, 0, stream>>>(maxb, -INFINITY, n);
    k_fill<<<fill_blk, TPB, 0, stream>>>(sumb, 0.0f, n);
    k_fill<<<fill_blk, TPB, 0, stream>>>(aggbuf, 0.0f, nh);
    k_edge_logit<<<edge_blk, TPB, 0, stream>>>(ni, nj, rscal, src, dst, et, logit, maxb, en);
    k_edge_exp<<<edge_blk, TPB, 0, stream>>>(logit, maxb, dst, expv, sumb, en);
    k_edge_alpha<<<edge_blk, TPB, 0, stream>>>(expv, sumb, dst, alpha1, alpha0, en);
    k_aggregate<<<edgew_blk, TPB, 0, stream>>>(hbuf, alpha1, src, dst, aggbuf, en);
    k_elu<<<fill_blk, TPB, 0, stream>>>(aggbuf, xbuf, nh);

    // ================= Head =================
    gemm(xbuf, pool_w, pool_b, pooled, n, HD, 1);   // relu(x @ pool_w + b)
    k_head<<<nodew_blk, TPB, 0, stream>>>(pooled, out_w, out_b, out, n);
}
